// TemporalRGCN_30236569764192
// MI455X (gfx1250) — compile-verified, matrix operand order checked by reading, not run on hardware
//
#include <hip/hip_runtime.h>

typedef _Float16 f16_t;
typedef __attribute__((ext_vector_type(16))) _Float16 v16h;
typedef __attribute__((ext_vector_type(8)))  _Float16 v8h;
typedef __attribute__((ext_vector_type(8)))  float    v8f;

#define H_DIM 128
#define R_REL 10
#define B_BAS 4
#define L_LAY 3
#define SA_STRIDE 40    // 16x32 f16 tile rows padded: 80B rows, 16B-aligned chunks
#define ST_STRIDE 136   // 128-wide f16 tile rows padded: 272B rows, 16B-aligned

// ---- WMMA helpers -----------------------------------------------------------
// A fragment (16x32 f16, M x K) per CDNA5 layout:
//   lanes 0-15 : row M=lane,    K = {0..7, 16..23}
//   lanes 16-31: row M=lane-16, K = {8..15, 24..31}
// B stored transposed in LDS as [N][K] so the same loader yields the B fragment.
__device__ __forceinline__ v16h load_frag(const _Float16* base, int stride, int lane) {
  const int m = lane & 15, half = lane >> 4;
  const _Float16* p = base + m * stride + half * 8;
  v8h a0 = *(const v8h*)(p);
  v8h a1 = *(const v8h*)(p + 16);
  return __builtin_shufflevector(a0, a1, 0,1,2,3,4,5,6,7,8,9,10,11,12,13,14,15);
}

__device__ __forceinline__ v8f wmma_f16(v16h a, v16h b, v8f c) {
  return __builtin_amdgcn_wmma_f32_16x16x32_f16(false, a, false, b, (short)0, c, false, false);
}

__device__ __forceinline__ void pack16(const float4& f0, const float4& f1,
                                       const float4& f2, const float4& f3,
                                       v8h& h0, v8h& h1) {
  h0[0]=(_Float16)f0.x; h0[1]=(_Float16)f0.y; h0[2]=(_Float16)f0.z; h0[3]=(_Float16)f0.w;
  h0[4]=(_Float16)f1.x; h0[5]=(_Float16)f1.y; h0[6]=(_Float16)f1.z; h0[7]=(_Float16)f1.w;
  h1[0]=(_Float16)f2.x; h1[1]=(_Float16)f2.y; h1[2]=(_Float16)f2.z; h1[3]=(_Float16)f2.w;
  h1[4]=(_Float16)f3.x; h1[5]=(_Float16)f3.y; h1[6]=(_Float16)f3.z; h1[7]=(_Float16)f3.w;
}

// ---- weight conversion (once per launch, trivial cost) ----------------------
// generic fp32 [batch][K][Ncols] -> f16 transposed [batch][Ncols][K]
__global__ void cvt_transpose_f16(const float* __restrict__ s, _Float16* __restrict__ d,
                                  int K, int Ncols, int total) {
  int i = blockIdx.x * blockDim.x + threadIdx.x;
  if (i >= total) return;
  int per = K * Ncols;
  int b = i / per, rem = i % per;
  int n = rem / K, k = rem % K;
  d[(size_t)b * per + (size_t)n * K + k] = (_Float16)s[(size_t)b * per + (size_t)k * Ncols + n];
}

// bases [L,B,H,H] -> WcatT [L][B*H][H]:  WcatT[l][b*H+j][f] = bases[l][b][f][j]
__global__ void cvt_bases_t(const float* __restrict__ s, _Float16* __restrict__ d) {
  int i = blockIdx.x * blockDim.x + threadIdx.x;
  const int per = B_BAS * H_DIM * H_DIM;
  if (i >= L_LAY * per) return;
  int l = i / per, rem = i % per;
  int col = rem / H_DIM, f = rem % H_DIM;
  int b = col >> 7, j = col & 127;
  d[i] = (_Float16)s[(((size_t)(l * B_BAS + b) * H_DIM) + f) * H_DIM + j];
}

// h2 [128][8] -> f16 transposed+padded [16][128]: d[n][k] = n<8 ? h2[k][n] : 0
__global__ void cvt_h2t(const float* __restrict__ s, _Float16* __restrict__ d) {
  int i = blockIdx.x * blockDim.x + threadIdx.x;
  if (i >= 16 * H_DIM) return;
  int n = i / H_DIM, k = i % H_DIM;
  d[i] = (_Float16)((n < 8) ? s[k * 8 + n] : 0.0f);
}

// ---- generic WMMA GEMM:  C[M,Ncols] = A[M,K](f32) @ W[K,Ncols] (+bias) ------
// W pre-transposed: Wt[Ncols][K] f16. block: 128 rows x 64 cols; 8 waves.
// k-loop software-pipelined: next tile's global loads issue during WMMA compute.
__global__ __launch_bounds__(256)
void gemm_bias_kernel(const float* __restrict__ A, const _Float16* __restrict__ Wt,
                      const float* __restrict__ bias,
                      float* __restrict__ outF, _Float16* __restrict__ outH,
                      int M, int K, int Ncols) {
  __shared__ _Float16 sA[128 * SA_STRIDE];
  __shared__ _Float16 sB[64 * SA_STRIDE];
  const int tid = threadIdx.x, lane = tid & 31, wave = tid >> 5;
  const int r0 = blockIdx.x * 128;
  const int n0 = blockIdx.y * 64;
  const int kIters = K >> 5;

  const int arow = tid >> 1, acg = (tid & 1) << 4;        // A: 128 rows x 32 k
  const int grow = r0 + arow;
  const int bn = tid >> 2, bkg = (tid & 3) << 3;          // B: 64 n x 32 k
  const float*    aptr = A  + (size_t)grow * K + acg;
  const _Float16* bptr = Wt + (size_t)(n0 + bn) * K + bkg;

  v8f acc[4] = {};
  float4 f0 = {0,0,0,0}, f1 = {0,0,0,0}, f2 = {0,0,0,0}, f3 = {0,0,0,0};
  v8h breg = {};
  if (grow < M) {
    const float4* ap = (const float4*)aptr;
    f0 = ap[0]; f1 = ap[1]; f2 = ap[2]; f3 = ap[3];
  }
  breg = *(const v8h*)bptr;

  for (int it = 0; it < kIters; ++it) {
    __syncthreads();                       // previous compute done: LDS reusable
    v8h h0, h1;
    pack16(f0, f1, f2, f3, h0, h1);
    *(v8h*)&sA[arow * SA_STRIDE + acg]     = h0;
    *(v8h*)&sA[arow * SA_STRIDE + acg + 8] = h1;
    *(v8h*)&sB[bn * SA_STRIDE + bkg]       = breg;
    if (it + 1 < kIters) {                 // prefetch next tile during compute
      if (grow < M) {
        const float4* ap = (const float4*)(aptr + (it + 1) * 32);
        f0 = ap[0]; f1 = ap[1]; f2 = ap[2]; f3 = ap[3];
      }
      breg = *(const v8h*)(bptr + (it + 1) * 32);
    }
    __syncthreads();
    v16h afrag = load_frag(&sA[(wave * 16) * SA_STRIDE], SA_STRIDE, lane);
#pragma unroll
    for (int t = 0; t < 4; ++t) {
      v16h bfrag = load_frag(&sB[(t * 16) * SA_STRIDE], SA_STRIDE, lane);
      acc[t] = wmma_f16(afrag, bfrag, acc[t]);
    }
  }
  const int half = lane >> 4, nn = lane & 15;
#pragma unroll
  for (int t = 0; t < 4; ++t) {
    int col = n0 + t * 16 + nn;
    float bv = bias ? bias[col] : 0.0f;
#pragma unroll
    for (int j = 0; j < 8; ++j) {
      int row = r0 + wave * 16 + half * 8 + j;
      if (row < M) {
        float v = acc[t][j] + bv;
        if (outF) outF[(size_t)row * Ncols + col] = v;
        else      outH[(size_t)row * Ncols + col] = (_Float16)v;
      }
    }
  }
}

// ---- per-(dst,rel) degree counts -------------------------------------------
__global__ void count_kernel(const int* __restrict__ ei, const int* __restrict__ et,
                             unsigned* __restrict__ cnt, int E) {
  int e = blockIdx.x * blockDim.x + threadIdx.x;
  if (e >= E) return;
  int dst = ei[E + e];
  int r   = et[e];
  atomicAdd(&cnt[(size_t)dst * R_REL + r], 1u);
}

// ---- edge messages: agg[dst] += w_e * sum_b comp[r,b] * xb[src,b,:] --------
__global__ void msg_kernel(const _Float16* __restrict__ xb, const int* __restrict__ ei,
                           const int* __restrict__ et, const float* __restrict__ comp_l,
                           const unsigned* __restrict__ cnt, float* __restrict__ agg, int E) {
  long long idx = (long long)blockIdx.x * blockDim.x + threadIdx.x;
  int e = (int)(idx >> 7);
  int h = (int)(idx & 127);
  if (e >= E) return;
  int src = ei[e], dst = ei[E + e], r = et[e];
  const float* c = comp_l + r * B_BAS;
  float w = 1.0f / fmaxf((float)cnt[(size_t)dst * R_REL + r], 1.0f);
  const _Float16* xr = xb + (size_t)src * (B_BAS * H_DIM);
  float m = c[0] * (float)xr[h]       + c[1] * (float)xr[128 + h]
          + c[2] * (float)xr[256 + h] + c[3] * (float)xr[384 + h];
  atomicAdd(&agg[(size_t)dst * H_DIM + h], m * w);
}

// ---- fused root GEMM + agg + bias + LayerNorm + ReLU + residual ------------
// block: 128 rows x full 128 cols; wave = 16 rows x 8 tiles -> LN in-register
__global__ __launch_bounds__(256)
void rgcn_root_ln_kernel(const float* __restrict__ Hin, const _Float16* __restrict__ WrootT,
                         const float* __restrict__ agg, const float* __restrict__ bias,
                         const float* __restrict__ g, const float* __restrict__ bln,
                         float* __restrict__ Hout, int M, int residual) {
  __shared__ _Float16 sA[128 * SA_STRIDE];
  __shared__ _Float16 sB[128 * SA_STRIDE];
  const int tid = threadIdx.x, lane = tid & 31, wave = tid >> 5;
  const int r0 = blockIdx.x * 128;

  const int arow = tid >> 1, acg = (tid & 1) << 4;
  const int grow = r0 + arow;
  const int bn = tid >> 1, bkg = (tid & 1) << 4;          // 128 n x 32 k
  const float*    aptr = Hin    + (size_t)grow * H_DIM + acg;
  const _Float16* bptr = WrootT + (size_t)bn * H_DIM + bkg;

  v8f acc[8] = {};
  float4 f0 = {0,0,0,0}, f1 = {0,0,0,0}, f2 = {0,0,0,0}, f3 = {0,0,0,0};
  v8h b0 = {}, b1 = {};
  if (grow < M) {
    const float4* ap = (const float4*)aptr;
    f0 = ap[0]; f1 = ap[1]; f2 = ap[2]; f3 = ap[3];
  }
  b0 = *(const v8h*)bptr;
  b1 = *(const v8h*)(bptr + 8);

  for (int it = 0; it < 4; ++it) {
    __syncthreads();
    v8h h0, h1;
    pack16(f0, f1, f2, f3, h0, h1);
    *(v8h*)&sA[arow * SA_STRIDE + acg]     = h0;
    *(v8h*)&sA[arow * SA_STRIDE + acg + 8] = h1;
    *(v8h*)&sB[bn * SA_STRIDE + bkg]       = b0;
    *(v8h*)&sB[bn * SA_STRIDE + bkg + 8]   = b1;
    if (it + 1 < 4) {
      if (grow < M) {
        const float4* ap = (const float4*)(aptr + (it + 1) * 32);
        f0 = ap[0]; f1 = ap[1]; f2 = ap[2]; f3 = ap[3];
      }
      b0 = *(const v8h*)(bptr + (it + 1) * 32);
      b1 = *(const v8h*)(bptr + (it + 1) * 32 + 8);
    }
    __syncthreads();
    v16h afrag = load_frag(&sA[(wave * 16) * SA_STRIDE], SA_STRIDE, lane);
#pragma unroll
    for (int t = 0; t < 8; ++t) {
      v16h bfrag = load_frag(&sB[(t * 16) * SA_STRIDE], SA_STRIDE, lane);
      acc[t] = wmma_f16(afrag, bfrag, acc[t]);
    }
  }
  const int half = lane >> 4, nn = lane & 15;
#pragma unroll
  for (int t = 0; t < 8; ++t) {
    int col = t * 16 + nn;
#pragma unroll
    for (int j = 0; j < 8; ++j) {
      int row = r0 + wave * 16 + half * 8 + j;
      float av = (row < M) ? agg[(size_t)row * H_DIM + col] : 0.0f;
      acc[t][j] += av + bias[col];
    }
  }
  // LayerNorm: row (wave,half,j) values live in-lane (8 tiles) x 16 lanes of this half
#pragma unroll
  for (int j = 0; j < 8; ++j) {
    float s = 0.0f, ss = 0.0f;
#pragma unroll
    for (int t = 0; t < 8; ++t) { float v = acc[t][j]; s += v; ss += v * v; }
#pragma unroll
    for (int off = 8; off >= 1; off >>= 1) {
      s  += __shfl_xor(s,  off, 16);
      ss += __shfl_xor(ss, off, 16);
    }
    float mean = s * (1.0f / 128.0f);
    float var  = ss * (1.0f / 128.0f) - mean * mean;
    float rstd = rsqrtf(var + 1e-5f);
    int row = r0 + wave * 16 + half * 8 + j;
    if (row < M) {
#pragma unroll
      for (int t = 0; t < 8; ++t) {
        int col = t * 16 + nn;
        float v = (acc[t][j] - mean) * rstd * g[col] + bln[col];
        v = fmaxf(v, 0.0f);
        if (residual) v += Hin[(size_t)row * H_DIM + col];
        Hout[(size_t)row * H_DIM + col] = v;
      }
    }
  }
}

// ---- fused head: out = relu(h@h1+b1) @ h2 + b2 -----------------------------
__global__ __launch_bounds__(256)
void head_kernel(const float* __restrict__ Hin, const _Float16* __restrict__ W1T,
                 const float* __restrict__ b1, const _Float16* __restrict__ W2T,
                 const float* __restrict__ b2, float* __restrict__ out, int M) {
  __shared__ _Float16 sA[128 * SA_STRIDE];
  __shared__ _Float16 sB[128 * SA_STRIDE];   // Bt1, then reused for Bt2
  __shared__ _Float16 sT[128 * ST_STRIDE];   // relu(h@h1+b1) tile, f16
  const int tid = threadIdx.x, lane = tid & 31, wave = tid >> 5;
  const int r0 = blockIdx.x * 128;
  const int half = lane >> 4, nn = lane & 15;

  const int arow = tid >> 1, acg = (tid & 1) << 4;
  const int grow = r0 + arow;
  const int bn = tid >> 1, bkg = (tid & 1) << 4;
  const float*    aptr = Hin + (size_t)grow * H_DIM + acg;
  const _Float16* bptr = W1T + (size_t)bn * H_DIM + bkg;

  v8f acc[8] = {};
  float4 f0 = {0,0,0,0}, f1 = {0,0,0,0}, f2 = {0,0,0,0}, f3 = {0,0,0,0};
  v8h b0 = {}, b1r = {};
  if (grow < M) {
    const float4* ap = (const float4*)aptr;
    f0 = ap[0]; f1 = ap[1]; f2 = ap[2]; f3 = ap[3];
  }
  b0  = *(const v8h*)bptr;
  b1r = *(const v8h*)(bptr + 8);

  for (int it = 0; it < 4; ++it) {
    __syncthreads();
    v8h h0, h1;
    pack16(f0, f1, f2, f3, h0, h1);
    *(v8h*)&sA[arow * SA_STRIDE + acg]     = h0;
    *(v8h*)&sA[arow * SA_STRIDE + acg + 8] = h1;
    *(v8h*)&sB[bn * SA_STRIDE + bkg]       = b0;
    *(v8h*)&sB[bn * SA_STRIDE + bkg + 8]   = b1r;
    if (it + 1 < 4) {
      if (grow < M) {
        const float4* ap = (const float4*)(aptr + (it + 1) * 32);
        f0 = ap[0]; f1 = ap[1]; f2 = ap[2]; f3 = ap[3];
      }
      b0  = *(const v8h*)(bptr + (it + 1) * 32);
      b1r = *(const v8h*)(bptr + (it + 1) * 32 + 8);
    }
    __syncthreads();
    v16h afrag = load_frag(&sA[(wave * 16) * SA_STRIDE], SA_STRIDE, lane);
#pragma unroll
    for (int t = 0; t < 8; ++t) {
      v16h bfrag = load_frag(&sB[(t * 16) * SA_STRIDE], SA_STRIDE, lane);
      acc[t] = wmma_f16(afrag, bfrag, acc[t]);
    }
  }
  __syncthreads();   // compute done; sB/sT free to overwrite
  // relu + bias -> sT (f16, A-operand of second GEMM)
#pragma unroll
  for (int t = 0; t < 8; ++t) {
    int col = t * 16 + nn;
#pragma unroll
    for (int j = 0; j < 8; ++j) {
      int rowl = wave * 16 + half * 8 + j;
      float v = (r0 + rowl < M) ? fmaxf(acc[t][j] + b1[col], 0.0f) : 0.0f;
      sT[rowl * ST_STRIDE + col] = (_Float16)v;
    }
  }
  { // Bt2: W2T is already [16][128]; copy with padded stride (v8h chunks)
    int n = tid >> 4, kg = (tid & 15) << 3;
    *(v8h*)&sB[n * ST_STRIDE + kg] = *(const v8h*)&W2T[(size_t)n * H_DIM + kg];
  }
  __syncthreads();
  v8f acc2 = {};
#pragma unroll
  for (int kk = 0; kk < 4; ++kk) {
    v16h a = load_frag(&sT[(wave * 16) * ST_STRIDE + kk * 32], ST_STRIDE, lane);
    v16h b = load_frag(&sB[kk * 32], ST_STRIDE, lane);
    acc2 = wmma_f16(a, b, acc2);
  }
  if (nn < 8) {
#pragma unroll
    for (int j = 0; j < 8; ++j) {
      int row = r0 + wave * 16 + half * 8 + j;
      if (row < M) out[(size_t)row * 8 + nn] = acc2[j] + b2[nn];
    }
  }
}

// ---- host orchestration -----------------------------------------------------
extern "C" void kernel_launch(void* const* d_in, const int* in_sizes, int n_in,
                              void* d_out, int out_size, void* d_ws, size_t ws_size,
                              hipStream_t stream) {
  (void)n_in; (void)out_size; (void)ws_size;
  const float* x      = (const float*)d_in[0];
  const int*   ei     = (const int*)d_in[1];
  const int*   et     = (const int*)d_in[2];
  const float* in_w   = (const float*)d_in[3];
  const float* in_b   = (const float*)d_in[4];
  const float* bases  = (const float*)d_in[5];
  const float* comp   = (const float*)d_in[6];
  const float* root   = (const float*)d_in[7];
  const float* conv_b = (const float*)d_in[8];
  const float* ln_g   = (const float*)d_in[9];
  const float* ln_b   = (const float*)d_in[10];
  const float* h1_w   = (const float*)d_in[11];
  const float* h1_b   = (const float*)d_in[12];
  const float* h2_w   = (const float*)d_in[13];
  const float* h2_b   = (const float*)d_in[14];

  const int N = in_sizes[0] / 64;   // F_IN = 64
  const int E = in_sizes[1] / 2;    // edge_index [2,E]

  // workspace carve (256B-aligned)
  char* p = (char*)d_ws;
  auto carve = [&](size_t bytes) -> char* {
    char* r = p; p += (bytes + 255) & ~(size_t)255; return r;
  };
  _Float16* wInWT  = (_Float16*)carve((size_t)128 * 64 * 2);            // [128 n][64 k]
  _Float16* wCatT  = (_Float16*)carve((size_t)L_LAY * 512 * 128 * 2);   // [l][512 n][128 k]
  _Float16* wRootT = (_Float16*)carve((size_t)L_LAY * 128 * 128 * 2);
  _Float16* wH1T   = (_Float16*)carve((size_t)128 * 128 * 2);
  _Float16* wH2T   = (_Float16*)carve((size_t)16 * 128 * 2);
  float*    hA     = (float*)carve((size_t)N * H_DIM * 4);
  float*    hB     = (float*)carve((size_t)N * H_DIM * 4);
  _Float16* xb     = (_Float16*)carve((size_t)N * (B_BAS * H_DIM) * 2); // ~102 MB, L2-resident
  float*    agg    = (float*)carve((size_t)N * H_DIM * 4);
  unsigned* cnt    = (unsigned*)carve((size_t)N * R_REL * 4);

  // weights -> f16, pre-transposed to [N][K] (once per launch)
  cvt_transpose_f16<<<(64 * 128 + 255) / 256, 256, 0, stream>>>(in_w, wInWT, 64, 128, 64 * 128);
  cvt_bases_t<<<(L_LAY * 512 * 128 + 255) / 256, 256, 0, stream>>>(bases, wCatT);
  cvt_transpose_f16<<<(L_LAY * 128 * 128 + 255) / 256, 256, 0, stream>>>(root, wRootT, 128, 128, L_LAY * 128 * 128);
  cvt_transpose_f16<<<(128 * 128 + 255) / 256, 256, 0, stream>>>(h1_w, wH1T, 128, 128, 128 * 128);
  cvt_h2t<<<(16 * 128 + 255) / 256, 256, 0, stream>>>(h2_w, wH2T);

  const int rowBlocks = (N + 127) / 128;

  // input projection: h = x @ in_w + in_b
  gemm_bias_kernel<<<dim3(rowBlocks, 2), 256, 0, stream>>>(
      x, wInWT, in_b, hA, nullptr, N, 64, 128);

  float* hcur = hA; float* hnext = hB;
  for (int l = 0; l < L_LAY; ++l) {
    // xb = h @ Wcat  (all 4 basis GEMMs fused, f16 output)
    gemm_bias_kernel<<<dim3(rowBlocks, 8), 256, 0, stream>>>(
        hcur, wCatT + (size_t)l * 512 * 128, nullptr, nullptr, xb, N, 128, 512);

    hipMemsetAsync(cnt, 0, (size_t)N * R_REL * 4, stream);
    hipMemsetAsync(agg, 0, (size_t)N * H_DIM * 4, stream);
    count_kernel<<<(E + 255) / 256, 256, 0, stream>>>(ei, et, cnt, E);

    long long tot = (long long)E * H_DIM;
    msg_kernel<<<(unsigned)((tot + 255) / 256), 256, 0, stream>>>(
        xb, ei, et, comp + (size_t)l * R_REL * B_BAS, cnt, agg, E);

    rgcn_root_ln_kernel<<<rowBlocks, 256, 0, stream>>>(
        hcur, wRootT + (size_t)l * 128 * 128, agg,
        conv_b + (size_t)l * H_DIM, ln_g + (size_t)l * H_DIM, ln_b + (size_t)l * H_DIM,
        hnext, N, (l > 0) ? 1 : 0);

    float* tmp = hcur; hcur = hnext; hnext = tmp;
  }

  head_kernel<<<rowBlocks, 256, 0, stream>>>(
      hcur, wH1T, h1_b, wH2T, h2_b, (float*)d_out, N);
}
